// Seq2SeqBridge_65944927863271
// MI455X (gfx1250) — compile-verified, matrix operand order checked by reading, not run on hardware
//
// Seq2SeqBridge on MI455X (gfx1250): WMMA-f16 LSTM steps, fp32 state math.
// - Weights packed once per launch into v_wmma_f32_16x16x32_f16 B-fragment
//   layout (f16, ~48MB -> L2-resident on 192MB L2).
// - x0 is constant => x0@Wih0^T + biases precomputed as a 4096-vector.
// - Per step: one fused kernel per LSTM layer; each wave owns a 16x16 M-tile
//   x 16 hidden units x all 4 gates, so sigmoid/tanh/cell update fuse into
//   the GEMM epilogue. h state is double-buffered in packed-A f16 layout.
// - B fragments explicitly double-buffered across kt; LDS staging uses the
//   CDNA5 async global->LDS path (global_load_async_to_lds_b128 + ASYNCcnt).
#include <hip/hip_runtime.h>

typedef __attribute__((ext_vector_type(16))) _Float16 v16h;
typedef __attribute__((ext_vector_type(8)))  float    v8f;

#define WMMA(a,b,c) __builtin_amdgcn_wmma_f32_16x16x32_f16(false,(a),false,(b),(short)0,(c),false,false)

#if defined(__gfx1250__) && __has_builtin(__builtin_amdgcn_global_load_async_to_lds_b128)
#define HAVE_ASYNC_LDS 1
// Builtin expects GCC-vector int4 pointers: addrspace(1) src, addrspace(3) dst.
typedef int i32x4 __attribute__((vector_size(16)));
typedef __attribute__((address_space(1))) i32x4 g_i32x4;
typedef __attribute__((address_space(3))) i32x4 l_i32x4;
#else
#define HAVE_ASYNC_LDS 0
#endif

__device__ __forceinline__ void wait_async_all() {
#if HAVE_ASYNC_LDS
#if __has_builtin(__builtin_amdgcn_s_wait_asynccnt)
  __builtin_amdgcn_s_wait_asynccnt(0);
#else
  asm volatile("s_wait_asynccnt 0x0" ::: "memory");
#endif
#endif
}

// Packed fragment layout for 16-bit WMMA operands (per CDNA5 ISA 7.12.2):
// lane<16 holds K' in {0..7,16..23}, lane>=16 holds K' in {8..15,24..31};
// fragment (mt,kt) is 512 halves, lane-contiguous 16-half groups -> b128 loads.
__device__ __forceinline__ int packed_off(int m, int k, int KT) {
  int mt = m >> 4, kt = k >> 5, kp = k & 31;
  int hi = (kp >> 3) & 1;
  int i  = (kp & 7) | ((kp >> 1) & 8);
  int lane = (m & 15) | (hi << 4);
  return ((mt * KT + kt) << 9) + (lane << 4) + i;
}

__device__ __forceinline__ float sigm(float x) { return 1.0f / (1.0f + __expf(-x)); }

// Stage a packed f16 state buffer into LDS. Async global->LDS (no VGPR
// round-trip, ASYNCcnt-tracked) when available; uint4 copy otherwise.
__device__ __forceinline__ void stage_lds(_Float16* dst, const _Float16* src, int nvec4) {
#if HAVE_ASYNC_LDS
  g_i32x4* g = (g_i32x4*)(uintptr_t)src;                     // generic->AS1: same value
  l_i32x4* l = (l_i32x4*)(uint32_t)(uintptr_t)dst;           // low 32 bits = LDS offset
  for (int i = threadIdx.x; i < nvec4; i += blockDim.x)
    __builtin_amdgcn_global_load_async_to_lds_b128(g + i, l + i, 0, 0);
  wait_async_all();
#else
  const uint4* s = (const uint4*)src;
  uint4* d = (uint4*)dst;
  for (int i = threadIdx.x; i < nvec4; i += blockDim.x) d[i] = s[i];
#endif
}

// Accumulate gates += A(sA, mt) @ W^T for 4 gate column-tiles.
// NT = column tiles per gate; B fragments double-buffered across kt.
template <int KT, int NT>
__device__ __forceinline__ void gemm_acc(v8f acc[4], const _Float16* sAbase,
                                         const _Float16* __restrict__ W, int jt, int mt,
                                         int lane) {
  auto bfrag = [&](int g, int kt) -> v16h {
    return *(const v16h*)(W + ((size_t)((g * NT + jt) * KT + kt) << 9) + (lane << 4));
  };
  v16h b0 = bfrag(0, 0), b1 = bfrag(1, 0), b2 = bfrag(2, 0), b3 = bfrag(3, 0);
  for (int kt = 0; kt < KT; ++kt) {
    v16h a = *(const v16h*)(sAbase + ((mt * KT + kt) << 9) + (lane << 4));
    if (kt + 2 < KT)
      __builtin_prefetch(W + ((size_t)(jt * KT + kt + 2) << 9), 0, 1);
    int kn = (kt + 1 < KT) ? kt + 1 : kt;
    v16h n0 = bfrag(0, kn), n1 = bfrag(1, kn), n2 = bfrag(2, kn), n3 = bfrag(3, kn);
    acc[0] = WMMA(a, b0, acc[0]);
    acc[1] = WMMA(a, b1, acc[1]);
    acc[2] = WMMA(a, b2, acc[2]);
    acc[3] = WMMA(a, b3, acc[3]);
    b0 = n0; b1 = n1; b2 = n2; b3 = n3;
  }
}

// ---------- one-time prep kernels ----------

// f32 row-major [N4][K] weight -> f16 packed B-fragments
__global__ void pack_w(const float* __restrict__ W, _Float16* __restrict__ o, int N4, int K) {
  int KT = K >> 5;
  int total = N4 * K;
  for (int idx = blockIdx.x * blockDim.x + threadIdx.x; idx < total;
       idx += gridDim.x * blockDim.x) {
    int n = idx / K, k = idx - n * K;
    o[packed_off(n, k, KT)] = (_Float16)W[idx];
  }
}

// pre0[n] = dot(emb_W[BOS], Wih0[n]) + bih0[n] + bhh0[n]   (batch-invariant)
__global__ void dec_pre0_k(const float* __restrict__ emb, const float* __restrict__ Wih0,
                           const float* __restrict__ bih, const float* __restrict__ bhh,
                           float* __restrict__ pre0) {
  int n = blockIdx.x * blockDim.x + threadIdx.x;
  if (n >= 4096) return;
  const float* x0 = emb + 512;  // BOS row (idx 1), E=512
  float s = 0.f;
  for (int e = 0; e < 512; ++e) s = fmaf(x0[e], Wih0[n * 512 + e], s);
  pre0[n] = s + bih[n] + bhh[n];
}

__global__ void add2(const float* a, const float* b, float* o, int n) {
  int i = blockIdx.x * blockDim.x + threadIdx.x;
  if (i < n) o[i] = a[i] + b[i];
}
__global__ void zero_h(_Float16* p, int n) {
  for (int i = blockIdx.x * blockDim.x + threadIdx.x; i < n; i += gridDim.x * blockDim.x)
    p[i] = (_Float16)0.f;
}
__global__ void zero_f(float* p, int n) {
  for (int i = blockIdx.x * blockDim.x + threadIdx.x; i < n; i += gridDim.x * blockDim.x)
    p[i] = 0.f;
}

__global__ void dec_init(const float* __restrict__ eh, const float* __restrict__ ec,
                         _Float16* h0pk0, _Float16* h1pk0, float* c0, float* c1) {
  int idx = blockIdx.x * blockDim.x + threadIdx.x;
  if (idx >= 32768) return;
  int b = idx >> 10, j = idx & 1023;
  h0pk0[packed_off(b, j, 32)] = (_Float16)eh[idx];
  h1pk0[packed_off(b, j, 32)] = (_Float16)eh[32768 + idx];
  c0[idx] = ec[idx];
  c1[idx] = ec[32768 + idx];
}

// ---------- decoder step kernels ----------
// grid 16 x 256thr (8 waves). wave gw -> hidden tile jt = gw>>1, M-tile mt = gw&1.

__global__ void dec_step_l0(const _Float16* __restrict__ h0pk_in, _Float16* __restrict__ h0pk_out,
                            float* __restrict__ c0, const _Float16* __restrict__ Whh0pk,
                            const float* __restrict__ pre0) {
  __shared__ alignas(32) _Float16 sA[32768];
  stage_lds(sA, h0pk_in, 4096);
  __syncthreads();
  int w = threadIdx.x >> 5, lane = threadIdx.x & 31;
  int gw = blockIdx.x * 8 + w, jt = gw >> 1, mt = gw & 1;
  v8f acc[4] = {{}, {}, {}, {}};
  gemm_acc<32, 64>(acc, sA, Whh0pk, jt, mt, lane);
  int j = (jt << 4) + (lane & 15), hi = lane >> 4;
  float pi = pre0[j], pf = pre0[1024 + j], pg = pre0[2048 + j], po = pre0[3072 + j];
#pragma unroll
  for (int r = 0; r < 8; ++r) {
    int b_ = mt * 16 + hi * 8 + r;
    float gi = acc[0][r] + pi, gf = acc[1][r] + pf;
    float gg = acc[2][r] + pg, go = acc[3][r] + po;
    float cold = c0[b_ * 1024 + j];
    float cn = sigm(gf) * cold + sigm(gi) * tanhf(gg);
    float hn = sigm(go) * tanhf(cn);
    c0[b_ * 1024 + j] = cn;
    h0pk_out[packed_off(b_, j, 32)] = (_Float16)hn;
  }
}

__global__ void dec_step_l1(const _Float16* __restrict__ h0pk, const _Float16* __restrict__ h1pk_in,
                            _Float16* __restrict__ h1pk_out, float* __restrict__ c1,
                            const _Float16* __restrict__ Wih1pk, const _Float16* __restrict__ Whh1pk,
                            const float* __restrict__ b1, float* __restrict__ seq_out, int t) {
  __shared__ alignas(32) _Float16 sA[32768];
  int w = threadIdx.x >> 5, lane = threadIdx.x & 31;
  int gw = blockIdx.x * 8 + w, jt = gw >> 1, mt = gw & 1;
  v8f acc[4] = {{}, {}, {}, {}};
  // phase 1: gates += h0 @ Wih1^T
  stage_lds(sA, h0pk, 4096);
  __syncthreads();
  gemm_acc<32, 64>(acc, sA, Wih1pk, jt, mt, lane);
  __syncthreads();
  // phase 2: gates += h1 @ Whh1^T (reuse LDS)
  stage_lds(sA, h1pk_in, 4096);
  __syncthreads();
  gemm_acc<32, 64>(acc, sA, Whh1pk, jt, mt, lane);
  int j = (jt << 4) + (lane & 15), hi = lane >> 4;
  float pi = b1[j], pf = b1[1024 + j], pg = b1[2048 + j], po = b1[3072 + j];
#pragma unroll
  for (int r = 0; r < 8; ++r) {
    int b_ = mt * 16 + hi * 8 + r;
    float gi = acc[0][r] + pi, gf = acc[1][r] + pf;
    float gg = acc[2][r] + pg, go = acc[3][r] + po;
    float cold = c1[b_ * 1024 + j];
    float cn = sigm(gf) * cold + sigm(gi) * tanhf(gg);
    float hn = sigm(go) * tanhf(cn);
    c1[b_ * 1024 + j] = cn;
    h1pk_out[packed_off(b_, j, 32)] = (_Float16)hn;
    seq_out[(size_t)b_ * 262144 + (size_t)t * 1024 + j] = hn;  // decoder output
  }
}

// ---------- encoder step kernel ----------
// grid (8, 2): blockIdx.y = direction; fwd/bwd run concurrently.
__global__ void enc_step(const float* __restrict__ X, float* __restrict__ Y,
                         const _Float16* __restrict__ WihB, const _Float16* __restrict__ WhhB,
                         const float* __restrict__ beB, const _Float16* __restrict__ hpkInB,
                         _Float16* __restrict__ hpkOutB, float* __restrict__ cB,
                         float* __restrict__ hFinB, int layer, int t) {
  const int Tn = 256;
  int dir = blockIdx.y;
  int tt = dir ? (Tn - 1 - t) : t;
  const _Float16* Wih = WihB + (size_t)(layer * 2 + dir) * (size_t)(128 * 32 * 512);
  const _Float16* Whh = WhhB + (size_t)(layer * 2 + dir) * (size_t)(128 * 16 * 512);
  const float* be = beB + (layer * 2 + dir) * 2048;
  const _Float16* hpkIn = hpkInB + dir * 16384;
  _Float16* hpkOut = hpkOutB + dir * 16384;
  float* cst = cB + dir * 16384;
  float* hfin = hFinB + dir * 16384;

  __shared__ alignas(32) _Float16 sA[32768];
  // pack x_t (f32 [32,1024]) into A-fragment f16 layout in LDS
  for (int idx = threadIdx.x; idx < 32768; idx += blockDim.x) {
    int m = idx >> 10, k = idx & 1023;
    sA[packed_off(m, k, 32)] = (_Float16)X[(size_t)m * 262144 + (size_t)tt * 1024 + k];
  }
  __syncthreads();
  int w = threadIdx.x >> 5, lane = threadIdx.x & 31;
  int gw = blockIdx.x * 8 + w, jt = gw >> 1, mt = gw & 1;  // jt in [0,32)
  v8f acc[4] = {{}, {}, {}, {}};
  // gates += x_t @ Wih^T  (K=1024)
  gemm_acc<32, 32>(acc, sA, Wih, jt, mt, lane);
  __syncthreads();
  stage_lds(sA, hpkIn, 2048);  // 32KB packed h
  __syncthreads();
  // gates += h @ Whh^T  (K=512)
  gemm_acc<16, 32>(acc, sA, Whh, jt, mt, lane);
  int j = (jt << 4) + (lane & 15), hi = lane >> 4;
  float pi = be[j], pf = be[512 + j], pg = be[1024 + j], po = be[1536 + j];
#pragma unroll
  for (int r = 0; r < 8; ++r) {
    int b_ = mt * 16 + hi * 8 + r;
    float gi = acc[0][r] + pi, gf = acc[1][r] + pf;
    float gg = acc[2][r] + pg, go = acc[3][r] + po;
    float cold = cst[b_ * 512 + j];
    float cn = sigm(gf) * cold + sigm(gi) * tanhf(gg);
    float hn = sigm(go) * tanhf(cn);
    cst[b_ * 512 + j] = cn;
    hfin[b_ * 512 + j] = hn;
    hpkOut[packed_off(b_, j, 16)] = (_Float16)hn;
    Y[(size_t)b_ * 262144 + (size_t)tt * 1024 + dir * 512 + j] = hn;
  }
}

// dec_h/dec_c = broadcast of concat(h_fwd,h_bwd) / concat(c_fwd,c_bwd) of last layer
__global__ void finalize_states(const float* __restrict__ hF, const float* __restrict__ cF,
                                float* __restrict__ o) {
  int idx = blockIdx.x * blockDim.x + threadIdx.x;
  if (idx >= 65536) return;
  int rem = idx & 32767, b = rem >> 10, j = rem & 1023;
  int dir = j >> 9, jj = j & 511;
  o[idx] = hF[dir * 16384 + b * 512 + jj];
  o[65536 + idx] = cF[dir * 16384 + b * 512 + jj];
}

// ---------- host driver ----------
extern "C" void kernel_launch(void* const* d_in, const int* in_sizes, int n_in,
                              void* d_out, int out_size, void* d_ws, size_t ws_size,
                              hipStream_t stream) {
  (void)in_sizes; (void)n_in; (void)out_size; (void)ws_size;
  const float* enc_h = (const float*)d_in[1];
  const float* enc_c = (const float*)d_in[2];
  const float* emb_W = (const float*)d_in[4];
  const float* dWih0 = (const float*)d_in[5];
  const float* dWhh0 = (const float*)d_in[6];
  const float* dbih0 = (const float*)d_in[7];
  const float* dbhh0 = (const float*)d_in[8];
  const float* dWih1 = (const float*)d_in[9];
  const float* dWhh1 = (const float*)d_in[10];
  const float* dbih1 = (const float*)d_in[11];
  const float* dbhh1 = (const float*)d_in[12];
  const float* eWih  = (const float*)d_in[13];
  const float* eWhh  = (const float*)d_in[14];
  const float* ebih  = (const float*)d_in[15];
  const float* ebhh  = (const float*)d_in[16];
  float* out = (float*)d_out;

  uint8_t* ws = (uint8_t*)d_ws;
  size_t off = 0;
  auto take = [&](size_t bytes) -> void* {
    void* p = ws + off;
    off = (off + bytes + 255) & ~(size_t)255;
    return p;
  };
  _Float16* pWhh0 = (_Float16*)take((size_t)4096 * 1024 * 2);
  _Float16* pWih1 = (_Float16*)take((size_t)4096 * 1024 * 2);
  _Float16* pWhh1 = (_Float16*)take((size_t)4096 * 1024 * 2);
  _Float16* pEih  = (_Float16*)take((size_t)4 * 2048 * 1024 * 2);
  _Float16* pEhh  = (_Float16*)take((size_t)4 * 2048 * 512 * 2);
  float* pre0 = (float*)take(4096 * 4);
  float* b1   = (float*)take(4096 * 4);
  float* beA  = (float*)take(8192 * 4);
  _Float16* h0pk = (_Float16*)take((size_t)2 * 32768 * 2);
  _Float16* h1pk = (_Float16*)take((size_t)2 * 32768 * 2);
  float* c0 = (float*)take(32768 * 4);
  float* c1 = (float*)take(32768 * 4);
  _Float16* ehpk = (_Float16*)take((size_t)2 * 2 * 16384 * 2);
  float* ecst  = (float*)take(2 * 16384 * 4);
  float* ehfin = (float*)take(2 * 16384 * 4);
  float* Yb = (float*)take((size_t)32 * 256 * 1024 * 4);

  // one-time (per launch) weight packing + constant-input precompute
  pack_w<<<2048, 256, 0, stream>>>(dWhh0, pWhh0, 4096, 1024);
  pack_w<<<2048, 256, 0, stream>>>(dWih1, pWih1, 4096, 1024);
  pack_w<<<2048, 256, 0, stream>>>(dWhh1, pWhh1, 4096, 1024);
  for (int l = 0; l < 2; ++l)
    for (int d = 0; d < 2; ++d) {
      pack_w<<<1024, 256, 0, stream>>>(eWih + (size_t)(l * 2 + d) * 2048 * 1024,
                                       pEih + (size_t)(l * 2 + d) * 2048 * 1024, 2048, 1024);
      pack_w<<<512, 256, 0, stream>>>(eWhh + (size_t)(l * 2 + d) * 2048 * 512,
                                      pEhh + (size_t)(l * 2 + d) * 2048 * 512, 2048, 512);
    }
  dec_pre0_k<<<16, 256, 0, stream>>>(emb_W, dWih0, dbih0, dbhh0, pre0);
  add2<<<16, 256, 0, stream>>>(dbih1, dbhh1, b1, 4096);
  add2<<<32, 256, 0, stream>>>(ebih, ebhh, beA, 8192);
  dec_init<<<128, 256, 0, stream>>>(enc_h, enc_c, h0pk, h1pk, c0, c1);

  // decoder: 256 sequential steps, 2 fused layer-kernels per step
  for (int t = 0; t < 256; ++t) {
    const _Float16* h0i = h0pk + (size_t)(t & 1) * 32768;
    _Float16* h0o = h0pk + (size_t)((t + 1) & 1) * 32768;
    const _Float16* h1i = h1pk + (size_t)(t & 1) * 32768;
    _Float16* h1o = h1pk + (size_t)((t + 1) & 1) * 32768;
    dec_step_l0<<<16, 256, 0, stream>>>(h0i, h0o, c0, pWhh0, pre0);
    dec_step_l1<<<16, 256, 0, stream>>>(h0o, h1i, h1o, c1, pWih1, pWhh1, b1, out, t);
  }

  // encoder: 2 layers, fwd+bwd concurrent via gridDim.y, ping-pong X/Y
  for (int l = 0; l < 2; ++l) {
    const float* X = (l == 0) ? (const float*)out : (const float*)Yb;
    float* Yl = (l == 0) ? Yb : out;
    zero_h<<<64, 256, 0, stream>>>(ehpk, 2 * 2 * 16384);
    zero_f<<<64, 256, 0, stream>>>(ecst, 2 * 16384);
    zero_f<<<64, 256, 0, stream>>>(ehfin, 2 * 16384);
    for (int t = 0; t < 256; ++t) {
      enc_step<<<dim3(8, 2), 256, 0, stream>>>(
          X, Yl, pEih, pEhh, beA,
          ehpk + (size_t)(t & 1) * 2 * 16384, ehpk + (size_t)((t + 1) & 1) * 2 * 16384,
          ecst, ehfin, l, t);
    }
  }
  finalize_states<<<256, 256, 0, stream>>>(ehfin, ecst, out + 8388608);
}
